// Attention_63952063038074
// MI455X (gfx1250) — matrix operand order repaired and checked by reading, not compiled, not run on hardware
//
#include <hip/hip_runtime.h>
#include <hip/hip_bf16.h>

// ---------------------------------------------------------------------------
// Types for CDNA5 WMMA (wave32): A/B = 16 bf16 in 8 VGPRs, C/D = 8 f32.
// ---------------------------------------------------------------------------
typedef __attribute__((ext_vector_type(16))) __bf16        v16bf;
typedef __attribute__((ext_vector_type(8)))  float         v8f;
typedef __attribute__((ext_vector_type(8)))  unsigned int  u32x8;
typedef __attribute__((ext_vector_type(4)))  unsigned int  u32x4;

union BfVec { u32x8 u; v16bf b; };

__device__ __forceinline__ unsigned short f2bf(float f) {
    unsigned int u = __float_as_uint(f);
    u += 0x7FFFu + ((u >> 16) & 1u);          // round-to-nearest-even
    return (unsigned short)(u >> 16);
}
__device__ __forceinline__ float bf2f(unsigned short s) {
    return __uint_as_float(((unsigned int)s) << 16);
}
__device__ __forceinline__ unsigned short to_bf16(float f)          { return f2bf(f); }
__device__ __forceinline__ unsigned short to_bf16(unsigned short s) { return s; }
__device__ __forceinline__ void store_out(unsigned short* p, float v) { *p = f2bf(v); }
__device__ __forceinline__ void store_out(float* p, float v)          { *p = v; }

__device__ __forceinline__ v8f wmma_bf16(const BfVec& a, const BfVec& b, v8f c) {
    return __builtin_amdgcn_wmma_f32_16x16x32_bf16(
        /*neg_a=*/false, a.b, /*neg_b=*/false, b.b,
        /*c_mod=*/(short)0, c, /*reuse_a=*/false, /*reuse_b=*/false);
}

// ---------------------------------------------------------------------------
// CDNA5 async global->LDS copy (ASYNCcnt path) via inline asm, per
// CDNA5 ISA / CDNA5 ISA:
//   GLOBAL_LOAD_ASYNC_TO_LDS_B128  vdst(LDS byte addr VGPR), vaddr(64b), off
// LDS byte offset = low 32 bits of the generic shared pointer (aperture map).
// Gated on gfx1250 so host / other targets compile the sync fallback.
// ---------------------------------------------------------------------------
#if defined(__gfx1250__)
#  define USE_ASYNC_LDS 1
#else
#  define USE_ASYNC_LDS 0
#endif

__device__ __forceinline__ void async_copy_b128(const void* gsrc, void* ldst) {
#if USE_ASYNC_LDS
    unsigned int       lds_off = (unsigned int)(unsigned long long)ldst;
    unsigned long long gaddr   = (unsigned long long)gsrc;
    asm volatile("global_load_async_to_lds_b128 %0, %1, off"
                 :: "v"(lds_off), "v"(gaddr) : "memory");
#else
    *(u32x4*)ldst = *(const u32x4*)gsrc;
#endif
}
__device__ __forceinline__ void async_wait() {
#if USE_ASYNC_LDS
    asm volatile("s_wait_asynccnt 0" ::: "memory");
#endif
}

// ---------------------------------------------------------------------------
// Tiled GEMM: C[M,N] = A[M,K] * B[K,N], f32->bf16 convert on stage, f32 acc.
// Block = 256 threads (8 waves). C tile 128x128; wave (w&3,w>>2) owns a 32x64
// sub-tile -> 8 WMMA per 32-deep k-step per wave. bf16-A tiles are staged via
// the CDNA5 async global->LDS path.
// Requires M%128==0, N%128==0, K%32==0 (true for all calls here).
// ---------------------------------------------------------------------------
template <typename AT, typename OT>
__global__ __launch_bounds__(256)
void gemm_bf16(const AT* __restrict__ A, const float* __restrict__ B,
               OT* __restrict__ C, int M, int Kd, int Nd)
{
    __shared__ __align__(32) unsigned short Asl[128][40];   // pitch 80B
    __shared__ __align__(32) unsigned short Bsl[32][144];   // pitch 288B (32B mult)

    const int m0 = blockIdx.y * 128;
    const int n0 = blockIdx.x * 128;
    const int tid  = threadIdx.x;
    const int w    = tid >> 5;          // 0..7
    const int lane = tid & 31;
    const int hi   = lane >> 4;         // wave32 lane-half
    const int nl   = lane & 15;
    const int rw   = (w & 3) * 32;      // wave row offset in C tile
    const int cw   = (w >> 2) * 64;     // wave col offset in C tile

    v8f acc[2][4] = {};

    for (int k0 = 0; k0 < Kd; k0 += 32) {
        if (k0 + 32 < Kd)
            __builtin_prefetch(&B[(size_t)(k0 + 32) * Nd + n0 + (tid & 127)], 0, 1);

        // ---- stage A (128x32) ----
        if constexpr (sizeof(AT) == 2) {
            // already bf16: async DMA 16B chunks straight into LDS
            for (int i = 0; i < 2; ++i) {
                int idx = tid + i * 256;             // 0..511 chunks
                int ra = idx >> 2, cj = (idx & 3) * 8;
                async_copy_b128(&A[(size_t)(m0 + ra) * Kd + k0 + cj], &Asl[ra][cj]);
            }
        } else {
            for (int i = 0; i < 16; ++i) {
                int idx = tid + i * 256;             // 0..4095
                int ra = idx >> 5, ca = idx & 31;
                Asl[ra][ca] = to_bf16(A[(size_t)(m0 + ra) * Kd + k0 + ca]);
            }
        }
        // ---- stage B (32x128), f32 -> bf16 ----
        for (int i = 0; i < 16; ++i) {
            int idx = tid + i * 256;                 // 0..4095
            int rb = idx >> 7, cb = idx & 127;
            Bsl[rb][cb] = f2bf(B[(size_t)(k0 + rb) * Nd + n0 + cb]);
        }
        async_wait();
        __syncthreads();

        // A fragments (rows rw+nl, rw+16+nl), permuted K pairs per lane-half.
        BfVec a0, a1, bfr[4];
#pragma unroll
        for (int vv = 0; vv < 8; ++vv) {
            int kb = ((vv >> 2) << 4) + (hi << 3) + ((vv & 3) << 1);
            a0.u[vv] = *(const unsigned int*)&Asl[rw + nl][kb];
            a1.u[vv] = *(const unsigned int*)&Asl[rw + 16 + nl][kb];
        }
#pragma unroll
        for (int f = 0; f < 4; ++f)
#pragma unroll
            for (int vv = 0; vv < 8; ++vv) {
                int kb = ((vv >> 2) << 4) + (hi << 3) + ((vv & 3) << 1);
                int cn = cw + f * 16 + nl;
                bfr[f].u[vv] = (unsigned int)Bsl[kb][cn] | ((unsigned int)Bsl[kb + 1][cn] << 16);
            }
#pragma unroll
        for (int f = 0; f < 4; ++f) {
            acc[0][f] = wmma_bf16(a0, bfr[f], acc[0][f]);
            acc[1][f] = wmma_bf16(a1, bfr[f], acc[1][f]);
        }
        __syncthreads();
    }

    // C layout: vgpr j -> row j + 8*hi; col = nl.
#pragma unroll
    for (int mi = 0; mi < 2; ++mi)
#pragma unroll
        for (int f = 0; f < 4; ++f)
#pragma unroll
            for (int j = 0; j < 8; ++j) {
                int r = m0 + rw + mi * 16 + j + hi * 8;
                int c = n0 + cw + f * 16 + nl;
                store_out(&C[(size_t)r * Nd + c], acc[mi][f][j]);
            }
}

// ---------------------------------------------------------------------------
// Split-ordering RoPE on bf16 buffer [T, heads, 128], in place.
// ---------------------------------------------------------------------------
__global__ void rope_kernel(unsigned short* buf, const int* __restrict__ pos,
                            int T, int heads)
{
    const int total = T * heads * 64;
    for (int idx = blockIdx.x * blockDim.x + threadIdx.x; idx < total;
         idx += gridDim.x * blockDim.x) {
        int j = idx & 63;
        int rest = idx >> 6;
        int h = rest % heads;
        int t = rest / heads;
        size_t base = (size_t)t * heads * 128 + (size_t)h * 128;
        float x1 = bf2f(buf[base + j]);
        float x2 = bf2f(buf[base + 64 + j]);
        // inv_freq = 10000^(-j/64) = exp(-j * ln(1e4)/64)
        float freq = __expf(-(float)j * (9.210340371976184f / 64.0f));
        float ang = (float)pos[t] * freq;
        float sn, cs;
        __sincosf(ang, &sn, &cs);
        buf[base + j]      = f2bf(x1 * cs - x2 * sn);
        buf[base + 64 + j] = f2bf(x2 * cs + x1 * sn);
    }
}

// ---------------------------------------------------------------------------
// Causal GQA flash attention, bf16 in/out, f32 online-softmax state.
// Block = (16-row q tile, kv head). 4 waves = the 4 grouped q heads, sharing
// one K^T / V LDS stage per 32-wide s-tile. V staged via async global->LDS.
// ---------------------------------------------------------------------------
__global__ __launch_bounds__(128)
void attn_kernel(const unsigned short* __restrict__ q,   // [T, 32*128] bf16
                 const unsigned short* __restrict__ k,   // [T,  8*128] bf16
                 const unsigned short* __restrict__ v,   // [T,  8*128] bf16
                 unsigned short* __restrict__ o,         // [T, 32*128] bf16
                 int T)
{
    const int t0  = blockIdx.x * 16;
    const int kh  = blockIdx.y;               // kv head 0..7
    const int tid = threadIdx.x;
    const int w   = tid >> 5;                 // group member g (0..3)
    const int lane = tid & 31;
    const int hi   = lane >> 4;
    const int nl   = lane & 15;
    const int head = kh * 4 + w;              // q head = kh*G + g

    __shared__ __align__(32) unsigned short KT[128][48];   // K^T [h][s_local], pitch 96B
    __shared__ __align__(32) unsigned short Vs[32][144];   // V [s_local][h],  pitch 288B
    __shared__ __align__(32) unsigned short Ps[4][16][36]; // per-wave P (16x32)

    // Q A-fragments for the 4 head-dim chunks (loaded once, reused all s-tiles)
    BfVec Qa[4];
    {
        const unsigned short* qbase = q + (size_t)(t0 + nl) * 4096 + head * 128;
#pragma unroll
        for (int c = 0; c < 4; ++c)
#pragma unroll
            for (int vv = 0; vv < 8; ++vv) {
                int kb = c * 32 + ((vv >> 2) << 4) + (hi << 3) + ((vv & 3) << 1);
                Qa[c].u[vv] = *(const unsigned int*)(qbase + kb);
            }
    }

    v8f O[8] = {};
    float mrow[8], lrow[8], alpha[8];
#pragma unroll
    for (int j = 0; j < 8; ++j) { mrow[j] = -1e30f; lrow[j] = 0.0f; }

    const float scale = 0.08838834764831845f;   // 1/sqrt(128)
    const int send = t0 + 16;                   // exclusive bound on s

    for (int s0 = 0; s0 < send; s0 += 32) {
        // K: b128 global chunks, transposed element-wise into KT.
        for (int i = 0; i < 4; ++i) {
            int idx = tid + i * 128;            // 0..511 chunks
            int sl = idx >> 4, cj = (idx & 15) * 8;
            union { u32x4 vv; unsigned short s[8]; } tmp;
            tmp.vv = *(const u32x4*)&k[(size_t)(s0 + sl) * 1024 + kh * 128 + cj];
#pragma unroll
            for (int e = 0; e < 8; ++e)
                KT[cj + e][sl] = tmp.s[e];
        }
        // V: async DMA straight into LDS (row-major, padded pitch).
        for (int i = 0; i < 4; ++i) {
            int idx = tid + i * 128;
            int sl = idx >> 4, cj = (idx & 15) * 8;
            async_copy_b128(&v[(size_t)(s0 + sl) * 1024 + kh * 128 + cj], &Vs[sl][cj]);
        }
        async_wait();
        __syncthreads();

        // Scores S[16 x 32] = Q (16x128) * K^T (128x32).
        v8f S0 = {}, S1 = {};
#pragma unroll
        for (int c = 0; c < 4; ++c) {
            BfVec b0, b1;
#pragma unroll
            for (int vv = 0; vv < 8; ++vv) {
                int kb = c * 32 + ((vv >> 2) << 4) + (hi << 3) + ((vv & 3) << 1);
                b0.u[vv] = (unsigned int)KT[kb][nl] | ((unsigned int)KT[kb + 1][nl] << 16);
                b1.u[vv] = (unsigned int)KT[kb][16 + nl] | ((unsigned int)KT[kb + 1][16 + nl] << 16);
            }
            S0 = wmma_bf16(Qa[c], b0, S0);
            S1 = wmma_bf16(Qa[c], b1, S1);
        }

        // Online softmax per row (rows live in 16-lane halves).
#pragma unroll
        for (int j = 0; j < 8; ++j) {
            int tg = t0 + j + hi * 8;
            float e0 = S0[j] * scale, e1 = S1[j] * scale;
            if (s0 + nl > tg)        e0 = -1e30f;   // causal mask
            if (s0 + 16 + nl > tg)   e1 = -1e30f;
            float mx = fmaxf(e0, e1);
#pragma unroll
            for (int off = 1; off < 16; off <<= 1)
                mx = fmaxf(mx, __shfl_xor(mx, off, 32));
            float mnew = fmaxf(mrow[j], mx);
            float a_   = __expf(mrow[j] - mnew);
            float p0 = __expf(e0 - mnew);
            float p1 = __expf(e1 - mnew);
            float rs = p0 + p1;
#pragma unroll
            for (int off = 1; off < 16; off <<= 1)
                rs += __shfl_xor(rs, off, 32);
            lrow[j]  = lrow[j] * a_ + rs;
            mrow[j]  = mnew;
            alpha[j] = a_;
            int pr = j + hi * 8;
            Ps[w][pr][nl]      = f2bf(p0);
            Ps[w][pr][16 + nl] = f2bf(p1);
        }
#pragma unroll
        for (int hf = 0; hf < 8; ++hf)
#pragma unroll
            for (int j = 0; j < 8; ++j)
                O[hf][j] *= alpha[j];

        // O += P (16x32) * V (32x128). P re-read from LDS in A layout.
        BfVec pa;
#pragma unroll
        for (int vv = 0; vv < 8; ++vv) {
            int kb = ((vv >> 2) << 4) + (hi << 3) + ((vv & 3) << 1);
            pa.u[vv] = *(const unsigned int*)&Ps[w][nl][kb];
        }
#pragma unroll
        for (int hf = 0; hf < 8; ++hf) {
            BfVec bv;
#pragma unroll
            for (int vv = 0; vv < 8; ++vv) {
                int kb = ((vv >> 2) << 4) + (hi << 3) + ((vv & 3) << 1);
                bv.u[vv] = (unsigned int)Vs[kb][hf * 16 + nl] | ((unsigned int)Vs[kb + 1][hf * 16 + nl] << 16);
            }
            O[hf] = wmma_bf16(pa, bv, O[hf]);
        }
        __syncthreads();
    }

    // Normalize and store attn output bf16 [T, 32*128].
#pragma unroll
    for (int hf = 0; hf < 8; ++hf)
#pragma unroll
        for (int j = 0; j < 8; ++j) {
            int tg = t0 + j + hi * 8;
            float val = O[hf][j] / lrow[j];
            o[(size_t)tg * 4096 + head * 128 + hf * 16 + nl] = f2bf(val);
        }
}

// ---------------------------------------------------------------------------
// Orchestration: QKV GEMMs -> RoPE(q,k) -> flash attention -> O-proj GEMM.
// Workspace: q(16MB) + k(4MB) + v(4MB) + attn(16MB) = 40MB bf16.
// ---------------------------------------------------------------------------
extern "C" void kernel_launch(void* const* d_in, const int* in_sizes, int n_in,
                              void* d_out, int out_size, void* d_ws, size_t ws_size,
                              hipStream_t stream)
{
    const float* x  = (const float*)d_in[0];
    const int*  pos = (const int*)  d_in[1];
    const float* wq = (const float*)d_in[2];
    const float* wk = (const float*)d_in[3];
    const float* wv = (const float*)d_in[4];
    const float* wo = (const float*)d_in[5];
    float* out = (float*)d_out;

    const int T = 2048, D = 4096, N = 32, Kh = 8, H = 128;

    unsigned short* qb = (unsigned short*)d_ws;            // [T, N*H]
    unsigned short* kb = qb + (size_t)T * N * H;           // [T, Kh*H]
    unsigned short* vb = kb + (size_t)T * Kh * H;          // [T, Kh*H]
    unsigned short* ab = vb + (size_t)T * Kh * H;          // [T, N*H]

    dim3 blk(256);
    // QKV projections (bf16 outputs)
    gemm_bf16<float, unsigned short>
        <<<dim3((N * H) / 128, T / 128), blk, 0, stream>>>(x, wq, qb, T, D, N * H);
    gemm_bf16<float, unsigned short>
        <<<dim3((Kh * H) / 128, T / 128), blk, 0, stream>>>(x, wk, kb, T, D, Kh * H);
    gemm_bf16<float, unsigned short>
        <<<dim3((Kh * H) / 128, T / 128), blk, 0, stream>>>(x, wv, vb, T, D, Kh * H);
    // RoPE in place on q and k
    rope_kernel<<<1024, 256, 0, stream>>>(qb, pos, T, N);
    rope_kernel<<<512,  256, 0, stream>>>(kb, pos, T, Kh);
    // Causal GQA attention
    attn_kernel<<<dim3(T / 16, Kh), dim3(128), 0, stream>>>(qb, kb, vb, ab, T);
    // Output projection (f32 output, bf16-A async-staged)
    gemm_bf16<unsigned short, float>
        <<<dim3(D / 128, T / 128), blk, 0, stream>>>(ab, wo, out, T, N * H, D);
}
